// ReliabilityTopKHead_30837865185700
// MI455X (gfx1250) — matrix operand order repaired and checked by reading, not compiled, not run on hardware
//
#include <hip/hip_runtime.h>
#include <hip/hip_bf16.h>

typedef __attribute__((ext_vector_type(16))) __bf16 v16bf;
typedef __attribute__((ext_vector_type(8)))  float  v8f;

#define B_    128
#define N_    4096
#define C_    768
#define H_    768
#define NCLS  1000
#define TOPK_ 16

#define KT_     (C_ / 32)  // 24 k-tiles (K=32 per WMMA)
#define NT_POOL (H_ / 16)  // 48 n-tiles
#define NT_FC   64         // 1024 padded columns (1000 real)

// ---- workspace layout (bytes) ----
#define WS_Z_OFF  0
#define WS_WB_OFF 393216
#define WS_FB_OFF (393216 + 1179648)

// TDM gather: builtin exists on both toolchains; arity differs.
//   clang < 23 (ROCm 7.2):        5 args (g0, g1, g2, g3, cpol)
//   clang >= 23 (amdgpu-toolchain): 6 args — extra int32x8 == the ISA's unused
//                                   VADDR4 SGPR group ("set to NULL"), zero-filled.
#if defined(__HIP_DEVICE_COMPILE__) && __has_builtin(__builtin_amdgcn_tensor_load_to_lds) && \
    __has_builtin(__builtin_amdgcn_s_wait_tensorcnt)
#define USE_TDM_GATHER 1
#else
#define USE_TDM_GATHER 0
#endif

// ---------------------------------------------------------------------------
// Swizzle pool_W_w and fc_w (f32, row-major [K][N]) into this file's (contiguous-k) WMMA B-fragment
// layout: frag[(nt*KT + kt)*32 + lane][e] = W[kt*32 + (lane>>4)*16 + e][nt*16 + (lane&15)]
// ---------------------------------------------------------------------------
__global__ __launch_bounds__(256) void prep_weights(const float* __restrict__ Ww,
                                                    const float* __restrict__ Fw,
                                                    unsigned short* __restrict__ Wb_,
                                                    unsigned short* __restrict__ Fb_) {
    __bf16* Wb = (__bf16*)Wb_;
    __bf16* Fb = (__bf16*)Fb_;
    const int gid = blockIdx.x * 256 + threadIdx.x;
    const int wLanes = NT_POOL * KT_ * 32;   // 36864
    const int fLanes = NT_FC   * KT_ * 32;   // 49152
    if (gid < wLanes) {
        const int lane = gid & 31;
        const int kt   = (gid >> 5) % KT_;
        const int nt   = (gid >> 5) / KT_;
        const int n    = nt * 16 + (lane & 15);
        const int kb   = kt * 32 + (lane >> 4) * 16;
        v16bf v;
#pragma unroll
        for (int e = 0; e < 16; ++e) v[e] = (__bf16)Ww[(size_t)(kb + e) * H_ + n];
        *(v16bf*)(Wb + (size_t)gid * 16) = v;
    } else if (gid < wLanes + fLanes) {
        const int g    = gid - wLanes;
        const int lane = g & 31;
        const int kt   = (g >> 5) % KT_;
        const int nt   = (g >> 5) / KT_;
        const int n    = nt * 16 + (lane & 15);
        const int kb   = kt * 32 + (lane >> 4) * 16;
        v16bf v;
#pragma unroll
        for (int e = 0; e < 16; ++e)
            v[e] = (n < NCLS) ? (__bf16)Fw[(size_t)(kb + e) * NCLS + n] : (__bf16)0.0f;
        *(v16bf*)(Fb + (size_t)g * 16) = v;
    }
}

// ---------------------------------------------------------------------------
// One block per batch: top-16 of r -> TDM gather-mode DMA of the 16 rows of x
// into LDS -> 16x768 @ 768x768 bf16 WMMA -> tanh/v score -> softmax -> z.
// ---------------------------------------------------------------------------
__global__ __launch_bounds__(256) void fused_pool(const float* __restrict__ x,
                                                  const float* __restrict__ r,
                                                  const unsigned short* __restrict__ Wb_,
                                                  const float* __restrict__ Wbias,
                                                  const float* __restrict__ vW,
                                                  float* __restrict__ z) {
    const __bf16* Wb = (const __bf16*)Wb_;
    __shared__ float s_xk[TOPK_ * C_];      // 48 KB; first 4096 floats reused for r
    __shared__ float s_red[256];
    __shared__ int   s_redi[256];
    __shared__ int   s_sel[TOPK_];
    __shared__ float s_score[TOPK_];
    __shared__ float s_alpha[TOPK_];

    const int tid = threadIdx.x;
    const int b   = blockIdx.x;

    // ---- 1. load r row into LDS ----
    for (int i = tid; i < N_; i += 256) s_xk[i] = r[(size_t)b * N_ + i];
    if (tid < TOPK_) s_score[tid] = 0.0f;
    __syncthreads();

    // ---- 2. iterative top-16 argmax ----
    for (int round = 0; round < TOPK_; ++round) {
        float best = -__builtin_inff();
        int   bi   = 0;
        for (int i = tid; i < N_; i += 256) {
            float v = s_xk[i];
            if (v > best) { best = v; bi = i; }
        }
        s_red[tid] = best; s_redi[tid] = bi;
        __syncthreads();
        for (int s = 128; s > 0; s >>= 1) {
            if (tid < s && s_red[tid + s] > s_red[tid]) {
                s_red[tid] = s_red[tid + s]; s_redi[tid] = s_redi[tid + s];
            }
            __syncthreads();
        }
        if (tid == 0) { s_sel[round] = s_redi[0]; s_xk[s_redi[0]] = -__builtin_inff(); }
        __syncthreads();
    }

    // ---- 3. gather top-k rows into LDS (overwrites r) ----
#if USE_TDM_GATHER
    if (tid < 32) {   // wave 0 issues one TDM gather-mode DMA for all 16 rows
        typedef unsigned int u32;
        typedef __attribute__((ext_vector_type(4))) u32 v4u;
        typedef __attribute__((ext_vector_type(8))) int v8i_;
        typedef __attribute__((ext_vector_type(4))) int v4i_;

        const unsigned long long gaddr =
            (unsigned long long)(const void*)(x + (size_t)b * N_ * C_);
        const u32 ldsa = (u32)(unsigned long long)&s_xk[0];  // low 32 bits = LDS offset

        v4u g0;
        g0[0] = 0x80000001u;                                  // count=1, gather_mode=1, 16b idx
        g0[1] = ldsa;                                         // lds_addr
        g0[2] = (u32)gaddr;                                   // global_addr[31:0]
        g0[3] = (u32)((gaddr >> 32) & 0x1FFFFFFu) | (2u << 30); // addr[56:32] | type=2

        v8i_ g1;
        g1[0] = (int)(2u << 16);                              // wg_mask=0, data_size=4B
        g1[1] = (int)(((u32)C_ & 0xFFFFu) << 16);             // tensor_dim0.lo (768)
        g1[2] = (int)((((u32)C_ >> 16) & 0xFFFFu) |
                      (((u32)N_ & 0xFFFFu) << 16));           // dim0.hi | tensor_dim1.lo (4096)
        g1[3] = (int)((((u32)N_ >> 16) & 0xFFFFu) |
                      ((u32)C_ << 16));                       // dim1.hi | tile_dim0 (768)
        g1[4] = TOPK_;                                        // tile_dim1 = #indices (16)
        g1[5] = C_;                                           // tensor_dim0_stride (768)
        g1[6] = 0;
        g1[7] = 0;

        int idxu[TOPK_];                                      // force SGPR-legal operands
#pragma unroll
        for (int k = 0; k < TOPK_; ++k)
            idxu[k] = __builtin_amdgcn_readfirstlane(s_sel[k]);

        v4i_ g2, g3;                                          // 16 x 16-bit row indices
#pragma unroll
        for (int p = 0; p < 4; ++p) {
            g2[p] = (idxu[2 * p]     & 0xFFFF) | (idxu[2 * p + 1] << 16);
            g3[p] = (idxu[8 + 2 * p] & 0xFFFF) | (idxu[8 + 2 * p + 1] << 16);
        }
#if __clang_major__ >= 23
        v8i_ g4 = {0, 0, 0, 0, 0, 0, 0, 0};                   // ISA VADDR4: unused, NULL
        __builtin_amdgcn_tensor_load_to_lds(g0, g1, g2, g3, g4, 0);
#else
        __builtin_amdgcn_tensor_load_to_lds(g0, g1, g2, g3, 0);
#endif
        __builtin_amdgcn_s_wait_tensorcnt(0);
    }
    __syncthreads();
#else
    for (int i = tid; i < TOPK_ * C_; i += 256) {
        const int kk = i / C_;
        const int c  = i % C_;
        s_xk[i] = x[((size_t)b * N_ + s_sel[kk]) * C_ + c];
    }
    __syncthreads();
#endif

    // ---- 4. pool GEMM: h = tanh(Xk @ W + b), score = h @ v ----
    const int lane = tid & 31;
    const int wave = tid >> 5;
    const int m    = lane & 15;   // A-matrix row for this lane
    const int kh   = lane >> 4;   // which K-half this lane holds

    v8f acc[6] = {};
    for (int kt = 0; kt < KT_; ++kt) {
        v16bf a;
        const float* xr = &s_xk[m * C_ + kt * 32 + kh * 16];
#pragma unroll
        for (int e = 0; e < 8; ++e) {
            a[e]     = (__bf16)xr[e];        // K = kt*32 + kh*16 + e
            a[8 + e] = (__bf16)xr[8 + e];    // K = kt*32 + kh*16 + 8 + e
        }
#pragma unroll
        for (int t = 0; t < 6; ++t) {
            const int nt = wave * 6 + t;
            v16bf bf = *(const v16bf*)(Wb + ((size_t)(nt * KT_ + kt) * 32 + lane) * 16);
            acc[t] = __builtin_amdgcn_wmma_f32_16x16x32_bf16(
                false, a, false, bf, (short)0, acc[t], false, false);
        }
    }

    // epilogue: tanh, dot with v, reduce per-row across lanes/waves
    float rowp[8];
#pragma unroll
    for (int i = 0; i < 8; ++i) rowp[i] = 0.0f;
#pragma unroll
    for (int t = 0; t < 6; ++t) {
        const int   n    = (wave * 6 + t) * 16 + (lane & 15);
        const float bias = Wbias[n];
        const float vv   = vW[n];
#pragma unroll
        for (int i = 0; i < 8; ++i) rowp[i] += tanhf(acc[t][i] + bias) * vv;
    }
    const int rbase = kh * 8;   // lanes 0-15 -> rows 0-7, lanes 16-31 -> rows 8-15
#pragma unroll
    for (int i = 0; i < 8; ++i) atomicAdd(&s_score[rbase + i], rowp[i]);
    __syncthreads();

    // ---- 5. softmax over 16 scores (pool_v_b shift is softmax-invariant) ----
    if (tid == 0) {
        float mx = -__builtin_inff();
        for (int k = 0; k < TOPK_; ++k) mx = fmaxf(mx, s_score[k]);
        float ssum = 0.0f;
        for (int k = 0; k < TOPK_; ++k) { float e = expf(s_score[k] - mx); s_alpha[k] = e; ssum += e; }
        const float inv = 1.0f / ssum;
        for (int k = 0; k < TOPK_; ++k) s_alpha[k] *= inv;
    }
    __syncthreads();

    // ---- 6. z = alpha^T * Xk (f32) ----
    for (int c = tid; c < C_; c += 256) {
        float zz = 0.0f;
#pragma unroll
        for (int k = 0; k < TOPK_; ++k) zz += s_alpha[k] * s_xk[k * C_ + c];
        z[(size_t)b * C_ + c] = zz;
    }
}

// ---------------------------------------------------------------------------
// logits = z @ fc_w + fc_b.  8 blocks (one 16-row M-tile each), 8 waves,
// each wave owns 8 N-tiles (N padded to 1024; stores guarded at 1000).
// ---------------------------------------------------------------------------
__global__ __launch_bounds__(256) void final_fc(const float* __restrict__ z,
                                                const unsigned short* __restrict__ Fb_,
                                                const float* __restrict__ fcb,
                                                float* __restrict__ out) {
    const __bf16* Fb = (const __bf16*)Fb_;
    __shared__ float s_z[16 * C_];   // 48 KB
    const int tid = threadIdx.x;
    const int mt  = blockIdx.x;

    for (int i = tid; i < 16 * C_; i += 256)
        s_z[i] = z[(size_t)(mt * 16 + i / C_) * C_ + (i % C_)];
    __syncthreads();

    const int lane = tid & 31;
    const int wave = tid >> 5;
    const int m    = lane & 15;
    const int kh   = lane >> 4;

    v8f acc[8] = {};
    for (int kt = 0; kt < KT_; ++kt) {
        v16bf a;
        const float* xr = &s_z[m * C_ + kt * 32 + kh * 16];
#pragma unroll
        for (int e = 0; e < 8; ++e) {
            a[e]     = (__bf16)xr[e];
            a[8 + e] = (__bf16)xr[8 + e];
        }
#pragma unroll
        for (int t = 0; t < 8; ++t) {
            const int nt = wave * 8 + t;
            v16bf bf = *(const v16bf*)(Fb + ((size_t)(nt * KT_ + kt) * 32 + lane) * 16);
            acc[t] = __builtin_amdgcn_wmma_f32_16x16x32_bf16(
                false, a, false, bf, (short)0, acc[t], false, false);
        }
    }

#pragma unroll
    for (int t = 0; t < 8; ++t) {
        const int n = (wave * 8 + t) * 16 + (lane & 15);
        if (n < NCLS) {
            const float bias = fcb[n];
#pragma unroll
            for (int i = 0; i < 8; ++i) {
                const int row = mt * 16 + kh * 8 + i;   // D: lanes<16 rows 0-7, >=16 rows 8-15
                out[(size_t)row * NCLS + n] = acc[t][i] + bias;
            }
        }
    }
}

// ---------------------------------------------------------------------------
extern "C" void kernel_launch(void* const* d_in, const int* in_sizes, int n_in,
                              void* d_out, int out_size, void* d_ws, size_t ws_size,
                              hipStream_t stream) {
    (void)in_sizes; (void)n_in; (void)out_size; (void)ws_size;
    const float* x     = (const float*)d_in[0];   // [B,N,C]
    const float* r     = (const float*)d_in[1];   // [B,N]
    const float* Ww    = (const float*)d_in[2];   // [C,H]
    const float* Wbias = (const float*)d_in[3];   // [H]
    const float* vw    = (const float*)d_in[4];   // [H,1]
    // d_in[5] = pool_v_b: constant shift, softmax-invariant -> unused
    const float* fw    = (const float*)d_in[6];   // [C,1000]
    const float* fcb   = (const float*)d_in[7];   // [1000]
    float* out = (float*)d_out;

    char* ws = (char*)d_ws;
    float*          zbuf = (float*)(ws + WS_Z_OFF);
    unsigned short* Wb   = (unsigned short*)(ws + WS_WB_OFF);
    unsigned short* Fb   = (unsigned short*)(ws + WS_FB_OFF);

    const int prepLanes = NT_POOL * KT_ * 32 + NT_FC * KT_ * 32;  // 86016
    prep_weights<<<prepLanes / 256, 256, 0, stream>>>(Ww, fw, Wb, Fb);
    fused_pool<<<B_, 256, 0, stream>>>(x, r, Wb, Wbias, vw, zbuf);
    final_fc<<<B_ / 16, 256, 0, stream>>>(zbuf, Fb, fcb, out);
}
